// VAE_52020643889438
// MI455X (gfx1250) — matrix operand order repaired and checked by reading, not compiled, not run on hardware
//
#include <hip/hip_runtime.h>
#include <hip/hip_bf16.h>
#include <math.h>
#include <stdint.h>

typedef __attribute__((ext_vector_type(16))) _Float16 v16h;
typedef __attribute__((ext_vector_type(8)))  _Float16 v8h;
typedef __attribute__((ext_vector_type(4)))  _Float16 v4h;
typedef __attribute__((ext_vector_type(8)))  float    v8f;

#define BS_   4096
#define D_IN_ 784
#define D_H_  400
#define D_Z_  20

#define BM 128
#define BN 64
#define BK 32
#define LDSK (BK + 8)   // padded row stride (halfs): 40*2B = 80B, multiple of 16B

union V16 { v16h v; v8h h[2]; };

__device__ __forceinline__ unsigned lds_off(const void* p) {
    // generic pointer to LDS: low 32 bits are the LDS byte offset
    return (unsigned)(uintptr_t)p;
}

__device__ __forceinline__ void async_ld_b128(unsigned lds, const void* gptr) {
    asm volatile("global_load_async_to_lds_b128 %0, %1, off"
                 :: "v"(lds), "v"((unsigned long long)(uintptr_t)gptr)
                 : "memory");
}

__device__ __forceinline__ void wait_async0() {
    asm volatile("s_wait_asynccnt 0" ::: "memory");
}

// C[M,N] = act(A[M,K] @ B[N,K]^T + bias[N]); A,B are f16, double-buffered async LDS staging
template<bool RELU, bool C_HALF>
__global__ __launch_bounds__(256)
void gemm_wmma_async(const _Float16* __restrict__ A, const _Float16* __restrict__ B,
                     const float* __restrict__ bias, void* __restrict__ Cv,
                     int M, int N, int K)
{
    __shared__ __align__(16) _Float16 As[2][BM][LDSK];
    __shared__ __align__(16) _Float16 Bs[2][BN][LDSK];

    const int tid  = threadIdx.x;
    const int lane = tid & 31;
    const int wid  = tid >> 5;
    const int wm   = (wid & 3) * 32;        // wave computes 32x32 of the 128x64 block tile
    const int wn   = (wid >> 2) * 32;
    const int m0   = blockIdx.x * BM;       // M = 4096, multiple of BM
    const int n0   = blockIdx.y * BN;
    const int halfw = lane >> 4;
    const int l16   = lane & 15;

    const int niter = (K + BK - 1) / BK;

    auto stage = [&](int buf, int k0) {
        const bool fullK = (k0 + BK <= K);
        // ---- A tile: BM x BK halfs = 8KB; 2 x 16B async chunks per thread ----
        if (fullK) {
            #pragma unroll
            for (int i = 0; i < 2; ++i) {
                int c = tid + 256 * i;
                int r = c >> 2, q = (c & 3) << 3;
                async_ld_b128(lds_off(&As[buf][r][q]),
                              A + (size_t)(m0 + r) * K + k0 + q);
            }
        } else {
            #pragma unroll
            for (int i = 0; i < 2; ++i) {
                int c = tid + 256 * i;
                int r = c >> 2, q = (c & 3) << 3;
                #pragma unroll
                for (int e = 0; e < 8; ++e) {
                    int k = k0 + q + e;
                    As[buf][r][q + e] = (k < K) ? A[(size_t)(m0 + r) * K + k]
                                                : (_Float16)0.f;
                }
            }
        }
        // ---- B tile: BN x BK halfs = 4KB; 1 x 16B async chunk per thread ----
        if (fullK && n0 + BN <= N) {
            int r = tid >> 2, q = (tid & 3) << 3;
            async_ld_b128(lds_off(&Bs[buf][r][q]),
                          B + (size_t)(n0 + r) * K + k0 + q);
        } else {
            int r = tid >> 2, q = (tid & 3) << 3;
            #pragma unroll
            for (int e = 0; e < 8; ++e) {
                int n = n0 + r, k = k0 + q + e;
                Bs[buf][r][q + e] = (n < N && k < K) ? B[(size_t)n * K + k]
                                                     : (_Float16)0.f;
            }
        }
    };

    stage(0, 0);
    wait_async0();
    __syncthreads();

    v8f acc[2][2] = {};

    for (int it = 0; it < niter; ++it) {
        const int buf = it & 1;
        if (it + 1 < niter) stage(buf ^ 1, (it + 1) * BK);

        // ---- fragment loads (CDNA5 wave32 VGPR layouts) + WMMA ----
        V16 afr[2], bfr[2];
        #pragma unroll
        for (int mt = 0; mt < 2; ++mt) {
            // A 16x32 f16: lane<16 -> K {0..7,16..23}; lane>=16 -> K {8..15,24..31}
            const _Float16* ar = &As[buf][wm + mt * 16 + l16][0];
            afr[mt].h[0] = *(const v8h*)(ar + halfw * 8);
            afr[mt].h[1] = *(const v8h*)(ar + 16 + halfw * 8);
        }
        #pragma unroll
        for (int nt = 0; nt < 2; ++nt) {
            // B 32x16 f16: lane<16 -> K {0..7,16..23}; lane>=16 -> K {8..15,24..31}
            const _Float16* br = &Bs[buf][wn + nt * 16 + l16][0];
            bfr[nt].h[0] = *(const v8h*)(br + halfw * 8);
            bfr[nt].h[1] = *(const v8h*)(br + 16 + halfw * 8);
        }
        #pragma unroll
        for (int mt = 0; mt < 2; ++mt)
            #pragma unroll
            for (int nt = 0; nt < 2; ++nt)
                acc[mt][nt] = __builtin_amdgcn_wmma_f32_16x16x32_f16(
                    false, afr[mt].v, false, bfr[nt].v,
                    (short)0, acc[mt][nt], false, false);

        if (it + 1 < niter) {          // uniform across block
            wait_async0();             // our async writes into buf^1 landed
            __syncthreads();           // everyone done reading buf, writes visible
        }
    }

    // ---- epilogue: C VGPR v -> row v (lanes<16) / v+8 (lanes>=16), col = lane%16 ----
    #pragma unroll
    for (int mt = 0; mt < 2; ++mt) {
        #pragma unroll
        for (int nt = 0; nt < 2; ++nt) {
            int col = n0 + wn + nt * 16 + l16;
            if (col >= N) continue;
            float bv = bias[col];
            #pragma unroll
            for (int v = 0; v < 8; ++v) {
                int row = m0 + wm + mt * 16 + v + halfw * 8;
                float val = acc[mt][nt][v] + bv;
                if (RELU) val = fmaxf(val, 0.f);
                if (C_HALF) ((_Float16*)Cv)[(size_t)row * N + col] = (_Float16)val;
                else        ((float*)Cv)[(size_t)row * N + col]   = val;
            }
        }
    }
}

// one-shot f32 -> f16 conversion (n must be a multiple of 4)
__global__ __launch_bounds__(256)
void cvt_f32_f16(const float* __restrict__ src, _Float16* __restrict__ dst, int n4)
{
    int i = blockIdx.x * blockDim.x + threadIdx.x;
    if (i >= n4) return;
    float4 f = ((const float4*)src)[i];
    v4h h = { (_Float16)f.x, (_Float16)f.y, (_Float16)f.z, (_Float16)f.w };
    ((v4h*)dst)[i] = h;
}

// mu/logvar heads: N=20, K=400 — one thread per (b, j), h1 read as f16
__global__ __launch_bounds__(256)
void head_kernel(const _Float16* __restrict__ h1,
                 const float* __restrict__ W21, const float* __restrict__ b21,
                 const float* __restrict__ W22, const float* __restrict__ b22,
                 float* __restrict__ mu, float* __restrict__ logvar)
{
    int idx = blockIdx.x * blockDim.x + threadIdx.x;
    if (idx >= BS_ * D_Z_) return;
    int b = idx / D_Z_, j = idx % D_Z_;
    const _Float16* hr = h1 + (size_t)b * D_H_;
    const float* w1 = W21 + (size_t)j * D_H_;
    const float* w2 = W22 + (size_t)j * D_H_;
    float s1 = 0.f, s2 = 0.f;
    #pragma unroll 4
    for (int k = 0; k < D_H_; ++k) {
        float h = (float)hr[k];
        s1 += h * w1[k];
        s2 += h * w2[k];
    }
    mu[idx]     = s1 + b21[j];
    logvar[idx] = s2 + b22[j];
}

// z[s,b,j] = mu[b,j] +/- sqrt(20)*logvar[b, s%20]*(j == s%20); emit z_last = z[39] as f16
__global__ __launch_bounds__(256)
void z_kernel(const float* __restrict__ mu, const float* __restrict__ logvar,
              float* __restrict__ z, _Float16* __restrict__ z_last)
{
    int idx = blockIdx.x * blockDim.x + threadIdx.x;
    if (idx >= 2 * D_Z_ * BS_ * D_Z_) return;
    int j = idx % D_Z_;
    int b = (idx / D_Z_) % BS_;
    int s = idx / (D_Z_ * BS_);
    int i = s % D_Z_;
    const float sqrtN = 4.47213595499958f;   // sqrt(20)
    float v = mu[(size_t)b * D_Z_ + j];
    if (j == i) {
        float d = sqrtN * logvar[(size_t)b * D_Z_ + i];
        v += (s < D_Z_) ? d : -d;
    }
    z[idx] = v;
    if (s == 2 * D_Z_ - 1) z_last[(size_t)b * D_Z_ + j] = (_Float16)v;
}

extern "C" void kernel_launch(void* const* d_in, const int* in_sizes, int n_in,
                              void* d_out, int out_size, void* d_ws, size_t ws_size,
                              hipStream_t stream)
{
    const float* x   = (const float*)d_in[0];
    const float* W1  = (const float*)d_in[1];
    const float* b1  = (const float*)d_in[2];
    const float* W21 = (const float*)d_in[3];
    const float* b21 = (const float*)d_in[4];
    const float* W22 = (const float*)d_in[5];
    const float* b22 = (const float*)d_in[6];
    const float* W3  = (const float*)d_in[7];
    const float* b3  = (const float*)d_in[8];
    const float* W41 = (const float*)d_in[9];
    const float* b41 = (const float*)d_in[10];
    const float* W42 = (const float*)d_in[11];
    const float* b42 = (const float*)d_in[12];

    // d_out layout: recon_mu | recon_var | mu | logvar | z
    float* out       = (float*)d_out;
    float* recon_mu  = out;
    float* recon_var = out + (size_t)BS_ * D_IN_;
    float* mu        = out + 2 * (size_t)BS_ * D_IN_;
    float* logvar    = mu + (size_t)BS_ * D_Z_;
    float* z         = logvar + (size_t)BS_ * D_Z_;

    // workspace: all-f16 operands (~14.4 MB)
    _Float16* w = (_Float16*)d_ws;
    _Float16* x_h   = w;  w += (size_t)BS_   * D_IN_;   // 3,211,264
    _Float16* W1_h  = w;  w += (size_t)D_H_  * D_IN_;   //   313,600
    _Float16* W3_h  = w;  w += (size_t)D_H_  * D_Z_;    //     8,000
    _Float16* W41_h = w;  w += (size_t)D_IN_ * D_H_;    //   313,600
    _Float16* W42_h = w;  w += (size_t)D_IN_ * D_H_;    //   313,600
    _Float16* h1    = w;  w += (size_t)BS_   * D_H_;    // 1,638,400
    _Float16* h3    = w;  w += (size_t)BS_   * D_H_;    // 1,638,400
    _Float16* zl_h  = w;                                //    81,920

    dim3 blk(256);

    // 0) one-shot f32->f16 conversion of GEMM operands
    cvt_f32_f16<<<(BS_*D_IN_/4 + 255)/256, blk, 0, stream>>>(x,   x_h,   BS_*D_IN_/4);
    cvt_f32_f16<<<(D_H_*D_IN_/4 + 255)/256, blk, 0, stream>>>(W1,  W1_h,  D_H_*D_IN_/4);
    cvt_f32_f16<<<(D_H_*D_Z_ /4 + 255)/256, blk, 0, stream>>>(W3,  W3_h,  D_H_*D_Z_/4);
    cvt_f32_f16<<<(D_IN_*D_H_/4 + 255)/256, blk, 0, stream>>>(W41, W41_h, D_IN_*D_H_/4);
    cvt_f32_f16<<<(D_IN_*D_H_/4 + 255)/256, blk, 0, stream>>>(W42, W42_h, D_IN_*D_H_/4);

    // 1) h1 = relu(x @ W1^T + b1)        M=4096 N=400 K=784
    gemm_wmma_async<true, true>
        <<<dim3(BS_/BM, (D_H_+BN-1)/BN), blk, 0, stream>>>(
            x_h, W1_h, b1, h1, BS_, D_H_, D_IN_);

    // 2) mu / logvar heads
    head_kernel<<<(BS_*D_Z_ + 255)/256, blk, 0, stream>>>(
        h1, W21, b21, W22, b22, mu, logvar);

    // 3) sigma-point expansion z + z_last (f16)
    z_kernel<<<(2*D_Z_*BS_*D_Z_ + 255)/256, blk, 0, stream>>>(mu, logvar, z, zl_h);

    // 4) h3 = relu(z_last @ W3^T + b3)   M=4096 N=400 K=20
    gemm_wmma_async<true, true>
        <<<dim3(BS_/BM, (D_H_+BN-1)/BN), blk, 0, stream>>>(
            zl_h, W3_h, b3, h3, BS_, D_H_, D_Z_);

    // 5) recon_mu  = h3 @ W41^T + b41    M=4096 N=784 K=400
    gemm_wmma_async<false, false>
        <<<dim3(BS_/BM, (D_IN_+BN-1)/BN), blk, 0, stream>>>(
            h3, W41_h, b41, recon_mu, BS_, D_IN_, D_H_);

    // 6) recon_var = h3 @ W42^T + b42    M=4096 N=784 K=400
    gemm_wmma_async<false, false>
        <<<dim3(BS_/BM, (D_IN_+BN-1)/BN), blk, 0, stream>>>(
            h3, W42_h, b42, recon_var, BS_, D_IN_, D_H_);

    (void)in_sizes; (void)n_in; (void)out_size; (void)ws_size;
}
